// MultiHeadAttention_33509334844034
// MI455X (gfx1250) — compile-verified, matrix operand order checked by reading, not run on hardware
//
#include <hip/hip_runtime.h>
#include <hip/hip_bf16.h>

typedef __attribute__((ext_vector_type(16))) _Float16 v16h;
typedef __attribute__((ext_vector_type(8)))  _Float16 v8h;
typedef __attribute__((ext_vector_type(8)))  float    v8f;
typedef __attribute__((ext_vector_type(4)))  float    v4f;

// ---------------------------------------------------------------------------
// WMMA wrapper: D = A(16x32 f16) * B(32x16 f16) + C(16x16 f32)
// ---------------------------------------------------------------------------
__device__ __forceinline__ v8f wmma16(v16h a, v16h b, v8f c) {
  return __builtin_amdgcn_wmma_f32_16x16x32_f16(
      /*neg_a=*/false, a, /*neg_b=*/false, b,
      /*c_mod=*/(short)0, c, /*reuse_a=*/false, /*reuse_b=*/false);
}

// ---------------------------------------------------------------------------
// Async global->LDS copy (CDNA5 ASYNCcnt path) + wait
// ---------------------------------------------------------------------------
__device__ __forceinline__ void async_ld16(const _Float16* g, const _Float16* l) {
  asm volatile("global_load_async_to_lds_b128 %0, %1, off"
               :: "v"((unsigned)(size_t)l), "v"(g)
               : "memory");
}

__device__ __forceinline__ void wait_async0() {
#if __has_builtin(__builtin_amdgcn_s_wait_asynccnt)
  __builtin_amdgcn_s_wait_asynccnt(0);
#else
  asm volatile("s_wait_asynccnt 0x0" ::: "memory");
#endif
}

// ---------------------------------------------------------------------------
// Operand tile loaders. Both A (16xK, lane = M) and B (Kx16, lane = N) use the
// same per-lane packing on CDNA5:
//   lane l, r = l&15, s = l>>4 : halfs[0..7]  = elem[row0+r][k0 + 8s .. +7]
//                                halfs[8..15] = elem[row0+r][k0 + 16 + 8s .. +7]
// ---------------------------------------------------------------------------
__device__ __forceinline__ v16h load_op_f32(const float* __restrict__ base,
                                            int ld, int row0, int k0) {
  const int l = threadIdx.x & 31;
  const int r = l & 15, s = l >> 4;
  const float* p0 = base + (size_t)(row0 + r) * ld + k0 + 8 * s;
  v4f x0 = ((const v4f*)p0)[0];
  v4f x1 = ((const v4f*)p0)[1];
  v4f x2 = ((const v4f*)(p0 + 16))[0];
  v4f x3 = ((const v4f*)(p0 + 16))[1];
  v16h o;
#pragma unroll
  for (int i = 0; i < 4; ++i) {
    o[i]      = (_Float16)x0[i];
    o[4 + i]  = (_Float16)x1[i];
    o[8 + i]  = (_Float16)x2[i];
    o[12 + i] = (_Float16)x3[i];
  }
  return o;
}

__device__ __forceinline__ v16h load_op_f16(const _Float16* base,
                                            int ld, int row0, int k0) {
  const int l = threadIdx.x & 31;
  const int r = l & 15, s = l >> 4;
  const _Float16* p0 = base + (size_t)(row0 + r) * ld + k0 + 8 * s;
  v8h c0 = *(const v8h*)p0;
  v8h c1 = *(const v8h*)(p0 + 16);
  v16h o;
#pragma unroll
  for (int i = 0; i < 8; ++i) { o[i] = c0[i]; o[8 + i] = c1[i]; }
  return o;
}

// ---------------------------------------------------------------------------
// One-shot f32 -> f16 weight conversion (weights are reused by ~128 waves;
// converting once halves B-operand bytes and removes cvt VALU from hot loops).
// n8 = elements/8.
// ---------------------------------------------------------------------------
__global__ __launch_bounds__(256) void cvt_kernel(const float* __restrict__ X,
                                                  _Float16* __restrict__ Y, int n8) {
  const int i = blockIdx.x * 256 + threadIdx.x;
  if (i >= n8) return;
  v4f x0 = ((const v4f*)X)[2 * i];
  v4f x1 = ((const v4f*)X)[2 * i + 1];
  v8h y;
#pragma unroll
  for (int j = 0; j < 4; ++j) { y[j] = (_Float16)x0[j]; y[4 + j] = (_Float16)x1[j]; }
  ((v8h*)Y)[i] = y;
}

// ---------------------------------------------------------------------------
// Projection: Y = X[4096,1024] * W^T + b  (W pre-converted to f16, [out,in]).
// mode 0: q -> f16 [B,H,S,64], scaled by 1/sqrt(64)
// mode 1: k -> f16 [B,H,S,64]
// mode 2: v -> f16 [B,H,64,S]  (transposed: attention B-operand contiguous)
// Block = 256 thr (8 waves); wave computes 32 rows x 64 cols; grid (128, 2).
// ---------------------------------------------------------------------------
__global__ __launch_bounds__(256) void proj_kernel(
    const float* __restrict__ X, const _Float16* __restrict__ Wh,
    const float* __restrict__ bias, _Float16* __restrict__ Y, int mode) {
  const int wid = threadIdx.x >> 5;
  const int l   = threadIdx.x & 31;
  const int m0  = blockIdx.x * 32;
  const int n0  = blockIdx.y * 512 + wid * 64;

  v8f acc[8] = {};
  for (int kk = 0; kk < 1024; kk += 32) {
    v16h a0 = load_op_f32(X, 1024, m0, kk);
    v16h a1 = load_op_f32(X, 1024, m0 + 16, kk);
#pragma unroll
    for (int t = 0; t < 4; ++t) {
      v16h b = load_op_f16(Wh, 1024, n0 + 16 * t, kk);
      acc[t]     = wmma16(a0, b, acc[t]);
      acc[4 + t] = wmma16(a1, b, acc[4 + t]);
    }
  }

  const int mrow = 8 * (l >> 4);
#pragma unroll
  for (int mi = 0; mi < 2; ++mi) {
#pragma unroll
    for (int t = 0; t < 4; ++t) {
      const int n = n0 + 16 * t + (l & 15);
      const float bv = bias[n];
      const int h = n >> 6, d = n & 63;
#pragma unroll
      for (int r = 0; r < 8; ++r) {
        const int m = m0 + 16 * mi + mrow + r;
        const int b = m >> 11, s = m & 2047;
        float v = acc[mi * 4 + t][r] + bv;
        if (mode == 0) v *= 0.125f;  // 1/sqrt(d_k)
        size_t idx;
        if (mode == 2) idx = ((size_t)((b * 16 + h) * 64 + d)) * 2048 + s;
        else           idx = ((size_t)((b * 16 + h) * 2048 + s)) * 64 + d;
        Y[idx] = (_Float16)v;
      }
    }
  }
}

// ---------------------------------------------------------------------------
// Fused flash attention. One wave owns a 16-query tile; all 8 waves of a block
// share the same (b,h), so K/V 32-key blocks are staged once per block into
// LDS via async global->LDS DMA, double buffered (load i+1 overlaps compute i).
// Scores computed TRANSPOSED (S' = K_blk * Q^T) so exp'd probabilities are
// already in A-operand layout for P*V. Online softmax; ctx f16 [B,S,H,64].
// ---------------------------------------------------------------------------
__global__ __launch_bounds__(256) void attn_kernel(
    const _Float16* __restrict__ Qh, const _Float16* __restrict__ Kh,
    const _Float16* __restrict__ Vt, _Float16* __restrict__ Ctx) {
  __shared__ _Float16 shK[2][32 * 64];  // [key][d]
  __shared__ _Float16 shV[2][64 * 32];  // [d][key]

  const int wid = threadIdx.x >> 5;
  const int l   = threadIdx.x & 31;
  const int t   = threadIdx.x;
  const int wg  = blockIdx.x * 8 + wid;   // 0..4095 wave-tiles
  const int bh  = wg >> 7;                // 0..31 (b*16+h); uniform per block
  const int q0  = (wg & 127) * 16;

  const _Float16* qp = Qh + (size_t)bh * 2048 * 64;
  const _Float16* kp = Kh + (size_t)bh * 2048 * 64;
  const _Float16* vp = Vt + (size_t)bh * 64 * 2048;

  // cooperative async stage of one 32-key block: 256 thr x 2 x 16B = 8KB
  const int krow = t >> 3, kcol = (t & 7) * 8;  // K: 32 rows x 64 d
  const int vrow = t >> 2, vcol = (t & 3) * 8;  // V: 64 d   x 32 keys

#define STAGE_KV(buf_, kb_)                                                     \
  do {                                                                          \
    async_ld16(kp + (size_t)((kb_) + krow) * 64 + kcol,                         \
               &shK[buf_][krow * 64 + kcol]);                                   \
    async_ld16(vp + (size_t)vrow * 2048 + (kb_) + vcol,                         \
               &shV[buf_][vrow * 32 + vcol]);                                   \
  } while (0)

  // Q^T held as B-operand for the whole loop (lane = query col, halfs = d)
  v16h qB0 = load_op_f16(qp, 64, q0, 0);
  v16h qB1 = load_op_f16(qp, 64, q0, 32);

  v8f o[4] = {v8f{}, v8f{}, v8f{}, v8f{}};
  float mmax = -3.0e38f, lsum = 0.0f;
  const int rowsel = (l >> 4) << 3;

  STAGE_KV(0, 0);
  int buf = 0;
  for (int kb = 0; kb < 2048; kb += 32) {
    wait_async0();      // this wave's stage-i DMA complete
    __syncthreads();    // everyone's stage-i visible; prior readers of buf^1 done
    if (kb + 32 < 2048) STAGE_KV(buf ^ 1, kb + 32);

    const _Float16* sk = shK[buf];
    const _Float16* sv = shV[buf];

    // S' tiles from LDS: M = key (relative), N = query
    v16h kA00 = load_op_f16(sk, 64, 0,  0);
    v16h kA01 = load_op_f16(sk, 64, 0,  32);
    v16h kA10 = load_op_f16(sk, 64, 16, 0);
    v16h kA11 = load_op_f16(sk, 64, 16, 32);
    v8f s0 = {}; s0 = wmma16(kA00, qB0, s0); s0 = wmma16(kA01, qB1, s0);
    v8f s1 = {}; s1 = wmma16(kA10, qB0, s1); s1 = wmma16(kA11, qB1, s1);

    // per-query (per-lane) block max; combine the two key-halves (l ^ 16)
    float bm = s0[0];
#pragma unroll
    for (int i = 0; i < 8; ++i) { bm = fmaxf(bm, s0[i]); bm = fmaxf(bm, s1[i]); }
    bm = fmaxf(bm, __shfl_xor(bm, 16, 32));
    const float mnew = fmaxf(mmax, bm);
    const float corr = __expf(mmax - mnew);
    mmax = mnew;

    // probabilities land directly in A-operand layout
    float ps = 0.0f;
    v16h pA;
#pragma unroll
    for (int i = 0; i < 8; ++i) {
      const float p0 = __expf(s0[i] - mnew);
      const float p1 = __expf(s1[i] - mnew);
      ps += p0 + p1;
      pA[i]     = (_Float16)p0;
      pA[8 + i] = (_Float16)p1;
    }
    ps += __shfl_xor(ps, 16, 32);
    lsum = lsum * corr + ps;

    // rescale O: row M = rowsel + r; its corr lives in lane (rowsel + r)
#pragma unroll
    for (int r = 0; r < 8; ++r) {
      const float cr = __shfl(corr, rowsel + r, 32);
#pragma unroll
      for (int tt = 0; tt < 4; ++tt) o[tt][r] *= cr;
    }

    // O += P(16x32) * V_block(32x64) from LDS
#pragma unroll
    for (int tt = 0; tt < 4; ++tt) {
      v16h vB = load_op_f16(sv, 32, 16 * tt, 0);
      o[tt] = wmma16(pA, vB, o[tt]);
    }
    buf ^= 1;
  }
#undef STAGE_KV

  // normalize and store ctx as [B, S, H, 64]
  const int b = bh >> 4, h = bh & 15;
#pragma unroll
  for (int r = 0; r < 8; ++r) {
    const float li = 1.0f / __shfl(lsum, rowsel + r, 32);
    const int s = q0 + rowsel + r;
#pragma unroll
    for (int tt = 0; tt < 4; ++tt) {
      const int d = 16 * tt + (l & 15);
      Ctx[((size_t)((b * 2048 + s) * 16 + h)) * 64 + d] = (_Float16)(o[tt][r] * li);
    }
  }
}

// ---------------------------------------------------------------------------
// Output projection: Out(f32) = Ctx(f16)[4096,1024] * Wo16^T + bo
// ---------------------------------------------------------------------------
__global__ __launch_bounds__(256) void outproj_kernel(
    const _Float16* __restrict__ Ctx, const _Float16* __restrict__ Wo16,
    const float* __restrict__ bo, float* __restrict__ Out) {
  const int wid = threadIdx.x >> 5;
  const int l   = threadIdx.x & 31;
  const int m0  = blockIdx.x * 32;
  const int n0  = blockIdx.y * 512 + wid * 64;

  v8f acc[8] = {};
  for (int kk = 0; kk < 1024; kk += 32) {
    v16h a0 = load_op_f16(Ctx, 1024, m0, kk);
    v16h a1 = load_op_f16(Ctx, 1024, m0 + 16, kk);
#pragma unroll
    for (int t = 0; t < 4; ++t) {
      v16h b = load_op_f16(Wo16, 1024, n0 + 16 * t, kk);
      acc[t]     = wmma16(a0, b, acc[t]);
      acc[4 + t] = wmma16(a1, b, acc[4 + t]);
    }
  }

  const int mrow = 8 * (l >> 4);
#pragma unroll
  for (int mi = 0; mi < 2; ++mi) {
#pragma unroll
    for (int t = 0; t < 4; ++t) {
      const int n = n0 + 16 * t + (l & 15);
      const float bv = bo[n];
#pragma unroll
      for (int r = 0; r < 8; ++r) {
        const int m = m0 + 16 * mi + mrow + r;
        Out[(size_t)m * 1024 + n] = acc[mi * 4 + t][r] + bv;
      }
    }
  }
}

// ---------------------------------------------------------------------------
// Host launcher
// ---------------------------------------------------------------------------
extern "C" void kernel_launch(void* const* d_in, const int* in_sizes, int n_in,
                              void* d_out, int out_size, void* d_ws, size_t ws_size,
                              hipStream_t stream) {
  const float* Q  = (const float*)d_in[0];
  const float* K  = (const float*)d_in[1];
  const float* V  = (const float*)d_in[2];
  const float* Wq = (const float*)d_in[3];
  const float* bq = (const float*)d_in[4];
  const float* Wk = (const float*)d_in[5];
  const float* bk = (const float*)d_in[6];
  const float* Wv = (const float*)d_in[7];
  const float* bv = (const float*)d_in[8];
  const float* Wo = (const float*)d_in[9];
  const float* bo = (const float*)d_in[10];

  const size_t HEAD_ELEMS = (size_t)2 * 16 * 2048 * 64;  // 4 Mi halfs = 8 MB
  const size_t W_ELEMS    = (size_t)1024 * 1024;
  _Float16* qbuf   = (_Float16*)d_ws;
  _Float16* kbuf   = qbuf + HEAD_ELEMS;
  _Float16* vtbuf  = kbuf + HEAD_ELEMS;
  _Float16* ctxbuf = vtbuf + HEAD_ELEMS;
  _Float16* wq16   = ctxbuf + HEAD_ELEMS;
  _Float16* wk16   = wq16 + W_ELEMS;
  _Float16* wv16   = wk16 + W_ELEMS;
  _Float16* wo16   = wv16 + W_ELEMS;

  dim3 blk(256);
  const int n8 = (int)(W_ELEMS / 8);          // 131072
  dim3 gcvt((n8 + 255) / 256);                // 512 blocks
  cvt_kernel<<<gcvt, blk, 0, stream>>>(Wq, wq16, n8);
  cvt_kernel<<<gcvt, blk, 0, stream>>>(Wk, wk16, n8);
  cvt_kernel<<<gcvt, blk, 0, stream>>>(Wv, wv16, n8);
  cvt_kernel<<<gcvt, blk, 0, stream>>>(Wo, wo16, n8);

  dim3 gproj(128, 2);  // 4096/32 x 1024/512
  proj_kernel<<<gproj, blk, 0, stream>>>(Q, wq16, bq, qbuf, 0);
  proj_kernel<<<gproj, blk, 0, stream>>>(K, wk16, bk, kbuf, 1);
  proj_kernel<<<gproj, blk, 0, stream>>>(V, wv16, bv, vtbuf, 2);
  attn_kernel<<<dim3(512), blk, 0, stream>>>(qbuf, kbuf, vtbuf, ctxbuf);
  outproj_kernel<<<gproj, blk, 0, stream>>>(ctxbuf, wo16, bo, (float*)d_out);
}
